// CL_26534307954983
// MI455X (gfx1250) — matrix operand order repaired and checked by reading, not compiled, not run on hardware
//
#include <hip/hip_runtime.h>
#include <hip/hip_bf16.h>

// ---------------------------------------------------------------------------
// CLIP-style patch-embed + symmetric contrastive loss for MI455X (gfx1250).
//   K0: one-shot weight f32->f16 conversion (weights are L2-resident and
//       reused by every workgroup; convert once, not per B-fragment)
//   K1: im2col-GEMM patch embed (f16 WMMA, f32 acc) + bias + L2-norm -> f16 E
//   K2: streaming logits GEMM with online logsumexp (run twice, A/B swapped)
// ---------------------------------------------------------------------------

typedef __attribute__((ext_vector_type(16))) _Float16 v16h;
typedef __attribute__((ext_vector_type(8)))  _Float16 v8h;
typedef __attribute__((ext_vector_type(4)))  _Float16 v4h;
typedef __attribute__((ext_vector_type(8)))  float    v8f;
typedef __attribute__((ext_vector_type(4)))  float    v4f;

#define BATCH 32
#define CIN   64
#define HW    128
#define PATCH 4
#define DIM   128
#define NPAT  1024                  // (128/4)^2 patches per image
#define MTOT  (BATCH * NPAT)        // 32768 patch rows per modality
#define KC    (CIN * PATCH * PATCH) // 1024 reduction dim of the conv-GEMM

// ---------------------------------------------------------------------------
// Kernel 0: convert both weight matrices [DIM][KC] f32 -> f16 once.
// grid = (DIM*KC/1024, 2), 256 threads, 4 elements/thread.
// ---------------------------------------------------------------------------
__global__ __launch_bounds__(256) void convert_w_kernel(
    const float* __restrict__ w1, const float* __restrict__ w2,
    _Float16* __restrict__ Wh)
{
  const int idx = blockIdx.x * 256 + threadIdx.x;    // index in float4 units
  const float* __restrict__ src = blockIdx.y ? w2 : w1;
  _Float16* __restrict__ dst = Wh + (size_t)blockIdx.y * ((size_t)DIM * KC);
  const v4f q = ((const v4f*)src)[idx];
  v4h h;
  h[0] = (_Float16)q.x; h[1] = (_Float16)q.y;
  h[2] = (_Float16)q.z; h[3] = (_Float16)q.w;
  ((v4h*)dst)[idx] = h;
}

// ---------------------------------------------------------------------------
// Kernel 1: patch embedding.
// grid = (256, 2): x -> 128-row tile of the 32768-row GEMM, y -> modality.
// block = 256 threads = 8 waves; wave w owns rows [tile*128 + w*16, +16) and
// all 128 output columns (8 WMMA col-tiles, 64 acc VGPRs).
// ---------------------------------------------------------------------------
__global__ __launch_bounds__(256) void embed_kernel(
    const float* __restrict__ imgs, const float* __restrict__ hha,
    const _Float16* __restrict__ Wh,
    const float* __restrict__ b1,   const float* __restrict__ b2,
    _Float16* __restrict__ Eout)
{
  const int wave = threadIdx.x >> 5;
  const int lane = threadIdx.x & 31;
  const int half = lane >> 4;    // WMMA lane half
  const int l15  = lane & 15;
  const int mod  = blockIdx.y;

  const float*    __restrict__ x  = mod ? hha : imgs;
  const _Float16* __restrict__ wh = Wh + (size_t)mod * ((size_t)DIM * KC);
  const float*    __restrict__ bi = mod ? b2 : b1;
  _Float16* __restrict__ E = Eout + (size_t)mod * ((size_t)MTOT * DIM);

  const int mbase = blockIdx.x * 128 + wave * 16;
  // A-matrix row handled by this lane (rows striped by l15 in WMMA A layout)
  const int m  = mbase + l15;
  const int b  = m >> 10;
  const int pt = m & 1023;
  const int ph = pt >> 5, pw = pt & 31;
  // NCHW base of this patch at channel 0, pixel row ph*4, col pw*4
  const float* xb = x + (((size_t)b * CIN) * HW + (size_t)ph * PATCH) * HW
                      + (size_t)pw * PATCH;

  v8f acc[8] = {};

  for (int k0 = 0; k0 < KC; k0 += 32) {
    // ---- A fragment: 16x32 f16. Lane holds two groups of 8 consecutive k.
    // k = c*16 + i*4 + j  ->  8 consecutive k (8-aligned) = channel c=kb/16,
    // pixel rows i0,i0+1 (i0 = (kb%16)/4), 4 contiguous floats each.
    v16h a;
    {
      int kb = k0 + half * 8;
      const float* p = xb + ((kb >> 4) << 14) + (((kb >> 2) & 3) << 7);
      v4f r0 = *(const v4f*)p;
      v4f r1 = *(const v4f*)(p + HW);
      a[0] = (_Float16)r0.x; a[1] = (_Float16)r0.y;
      a[2] = (_Float16)r0.z; a[3] = (_Float16)r0.w;
      a[4] = (_Float16)r1.x; a[5] = (_Float16)r1.y;
      a[6] = (_Float16)r1.z; a[7] = (_Float16)r1.w;

      kb = k0 + 16 + half * 8;
      p = xb + ((kb >> 4) << 14) + (((kb >> 2) & 3) << 7);
      r0 = *(const v4f*)p;
      r1 = *(const v4f*)(p + HW);
      a[8]  = (_Float16)r0.x; a[9]  = (_Float16)r0.y;
      a[10] = (_Float16)r0.z; a[11] = (_Float16)r0.w;
      a[12] = (_Float16)r1.x; a[13] = (_Float16)r1.y;
      a[14] = (_Float16)r1.z; a[15] = (_Float16)r1.w;
    }

    // ---- 8 column tiles: B fragment = two groups of 8 consecutive K of wh[col][*] (same k order as A)
#pragma unroll
    for (int t = 0; t < 8; ++t) {
      const _Float16* wp = wh + (size_t)(t * 16 + l15) * KC + k0 + half * 8;
      const v8h lo = ((const v8h*)wp)[0];
      const v8h hi = ((const v8h*)wp)[2];
      v16h bf;
#pragma unroll
      for (int i = 0; i < 8; ++i) { bf[i] = lo[i]; bf[8 + i] = hi[i]; }

      acc[t] = __builtin_amdgcn_wmma_f32_16x16x32_f16(
          false, a, false, bf, (short)0, acc[t], false, false);
    }
  }

  // ---- bias, per-row L2 normalization, store f16 embeddings row-major [m][d]
#pragma unroll
  for (int t = 0; t < 8; ++t) {
    const float bv = bi[t * 16 + l15];     // C/D layout: col = t*16 + l15
#pragma unroll
    for (int r = 0; r < 8; ++r) acc[t][r] += bv;
  }

#pragma unroll
  for (int r = 0; r < 8; ++r) {
    // row for this (r, half) pair = mbase + r + 8*half
    float s = 0.f;
#pragma unroll
    for (int t = 0; t < 8; ++t) s += acc[t][r] * acc[t][r];
    s += __shfl_xor(s, 1);  s += __shfl_xor(s, 2);
    s += __shfl_xor(s, 4);  s += __shfl_xor(s, 8);   // stays inside each half
    const float inv = rsqrtf(s);
    const int mrow = mbase + r + half * 8;
#pragma unroll
    for (int t = 0; t < 8; ++t) {
      E[(size_t)mrow * DIM + t * 16 + l15] = (_Float16)(acc[t][r] * inv);
    }
  }
}

// ---------------------------------------------------------------------------
// Kernel 2: S = scale * EA @ EB^T streamed in 16-col tiles with online LSE.
// grid = (8 row-blocks, 32 batches), 8 waves/block, 16 rows/wave.
// Adds sum(rowLSE - diag) * wloss into *out.  Called twice (A/B swapped)
// so the two passes give (loss_i + loss_h) * 0.5.
// ---------------------------------------------------------------------------
__global__ __launch_bounds__(256) void clip_loss_kernel(
    const _Float16* __restrict__ EA, const _Float16* __restrict__ EB,
    const float* __restrict__ logit_scale, float* __restrict__ out,
    float wloss)
{
  const int wave = threadIdx.x >> 5;
  const int lane = threadIdx.x & 31;
  const int half = lane >> 4;
  const int l15  = lane & 15;
  const int b    = blockIdx.y;
  const int rowbase = blockIdx.x * 128 + wave * 16;
  const float scale = __expf(logit_scale[0]);

  // Preload all 4 A fragments (K = 128 split into 4 steps of 32)
  const _Float16* arow = EA + ((size_t)b * NPAT + rowbase + l15) * DIM;
  v16h afr[4];
#pragma unroll
  for (int ks = 0; ks < 4; ++ks) {
    const int d0 = ks * 32;
    v8h lo = *(const v8h*)(arow + d0 + half * 8);
    v8h hi = *(const v8h*)(arow + d0 + 16 + half * 8);
#pragma unroll
    for (int i = 0; i < 8; ++i) { afr[ks][i] = lo[i]; afr[ks][8 + i] = hi[i]; }
  }

  float mrun[8], srun[8];
#pragma unroll
  for (int r = 0; r < 8; ++r) { mrun[r] = -3.0e38f; srun[r] = 0.f; }
  float dacc = 0.f;
  const int diag_jt = rowbase >> 4;   // col-tile containing this wave's diagonal

  for (int jt = 0; jt < 64; ++jt) {
    const _Float16* bcol =
        EB + ((size_t)b * NPAT + jt * 16 + l15) * DIM + half * 8;
    v8f acc = {};
#pragma unroll
    for (int ks = 0; ks < 4; ++ks) {
      v8h lo = *(const v8h*)(bcol + ks * 32);
      v8h hi = *(const v8h*)(bcol + ks * 32 + 16);
      v16h bf;
#pragma unroll
      for (int i = 0; i < 8; ++i) { bf[i] = lo[i]; bf[8 + i] = hi[i]; }
      acc = __builtin_amdgcn_wmma_f32_16x16x32_f16(
          false, afr[ks], false, bf, (short)0, acc, false, false);
    }
    // per-lane online logsumexp over this lane's column subset
#pragma unroll
    for (int r = 0; r < 8; ++r) {
      const float v = scale * acc[r];
      // diagonal element (row == col) lives at lane l15 == r + 8*half
      dacc += (jt == diag_jt && l15 == r + 8 * half) ? v : 0.f;
      const float mn = fmaxf(mrun[r], v);
      srun[r] = srun[r] * __expf(mrun[r] - mn) + __expf(v - mn);
      mrun[r] = mn;
    }
  }

  // merge (m, s) pairs across the 16 lanes of each half, pick one lane per row
  float partial = -dacc;
#pragma unroll
  for (int r = 0; r < 8; ++r) {
    float mr = mrun[r], sr = srun[r];
#pragma unroll
    for (int mask = 1; mask <= 8; mask <<= 1) {
      const float mo = __shfl_xor(mr, mask);
      const float so = __shfl_xor(sr, mask);
      const float mn = fmaxf(mr, mo);
      sr = sr * __expf(mr - mn) + so * __expf(mo - mn);
      mr = mn;
    }
    const float lse = mr + __logf(sr);
    partial += (l15 == r + 8 * half) ? lse : 0.f;   // one lane per row
  }
#pragma unroll
  for (int mask = 1; mask <= 16; mask <<= 1) partial += __shfl_xor(partial, mask);
  if (lane == 0) atomicAdd(out, partial * wloss);
}

// ---------------------------------------------------------------------------
extern "C" void kernel_launch(void* const* d_in, const int* in_sizes, int n_in,
                              void* d_out, int out_size, void* d_ws,
                              size_t ws_size, hipStream_t stream)
{
  const float* imgs = (const float*)d_in[0];
  const float* hha  = (const float*)d_in[1];
  const float* w1   = (const float*)d_in[2];
  const float* b1   = (const float*)d_in[3];
  const float* w2   = (const float*)d_in[4];
  const float* b2   = (const float*)d_in[5];
  const float* ls   = (const float*)d_in[6];
  float* out = (float*)d_out;

  // workspace: embeddings (2 x 32768 x 128 f16 = 16 MB) then f16 weights (512 KB)
  _Float16* E    = (_Float16*)d_ws;
  _Float16* Eimg = E;
  _Float16* Ehha = E + (size_t)MTOT * DIM;
  _Float16* Wh   = E + 2 * (size_t)MTOT * DIM;

  hipMemsetAsync(out, 0, sizeof(float), stream);

  dim3 blk(256);
  dim3 g0((DIM * KC) / (256 * 4), 2);
  convert_w_kernel<<<g0, blk, 0, stream>>>(w1, w2, Wh);

  dim3 g1(256, 2);
  embed_kernel<<<g1, blk, 0, stream>>>(imgs, hha, Wh, b1, b2, E);

  const float wloss = 0.5f / (float)MTOT;
  dim3 g2(8, BATCH);
  // pass 1: row-LSE of S (image -> hha), pass 2: row-LSE of S^T == col-LSE
  clip_loss_kernel<<<g2, blk, 0, stream>>>(Eimg, Ehha, ls, out, wloss);
  clip_loss_kernel<<<g2, blk, 0, stream>>>(Ehha, Eimg, ls, out, wloss);
}
